// encoder_layer_22016002360042
// MI455X (gfx1250) — compile-verified, matrix operand order checked by reading, not run on hardware
//
#include <hip/hip_runtime.h>
#include <hip/hip_bf16.h>

// ---------------------------------------------------------------------------
// Types for CDNA5 WMMA (wave32): A/B = 16 halfs (8 VGPRs), C/D = 8 floats.
// ---------------------------------------------------------------------------
typedef __attribute__((ext_vector_type(16))) _Float16 v16h;
typedef __attribute__((ext_vector_type(8)))  _Float16 v8h;
typedef __attribute__((ext_vector_type(8)))  float    v8f;

// Problem constants (fixed by the reference).
#define BB 32
#define SS 512
#define DD 512
#define HH 8
#define DK 64
#define FFP 256   // FF=200 padded to 256
#define FF  200
#define MTOT (BB*SS)            // 16384 rows
#define LN_N (SS*DD)            // 262144 elements per batch

// ---------------------------------------------------------------------------
// Fragment loaders for 16-bit WMMA operands, row-major storage.
//   A-matrix 16x32 layout: lanes 0-15 hold row M=lane, K in {kk..kk+7, kk+16..kk+23}
//                          lanes 16-31 hold row M=lane-16, K in {kk+8..kk+15, kk+24..kk+31}
//   B-matrix (N x K "weight" layout) is loaded identically with N in place of M.
// ---------------------------------------------------------------------------
__device__ inline v16h frag_at(const _Float16* __restrict__ p, int kk) {
    // p must already include the row offset and the +hi*8 lane shift.
    v8h lo = *(const v8h*)(p + kk);
    v8h hv = *(const v8h*)(p + kk + 16);
    return __builtin_shufflevector(lo, hv, 0,1,2,3,4,5,6,7,8,9,10,11,12,13,14,15);
}

__device__ inline v16h load_frag(const _Float16* __restrict__ base, int ld,
                                 int row, int kk, int hi) {
    return frag_at(base + (size_t)row * ld + hi * 8, kk);
}

__device__ inline v8f wmma_f16(v16h a, v16h b, v8f c) {
    return __builtin_amdgcn_wmma_f32_16x16x32_f16(false, a, false, b,
                                                  (short)0, c, false, false);
}

// ---------------------------------------------------------------------------
// Generic WMMA GEMM: C[M,N] = A[M,K] * B[N,K]^T (+bias) (+relu) (+residual)
// A row-major f16 (ld=K), B = torch Linear weight layout [out=N, in=K] f16.
// Each wave computes a 32x64 strip (2 A-frags x 4 B-frags = 8 WMMAs/K-step);
// block = 4 waves = 128x64 tile.
//
// The K-loop is unrolled 2x with explicit ping-pong fragment slots: slot B for
// kk+32 is loaded while slot A's WMMAs issue and vice versa. Unlike a rotating
// double buffer (a = na), this needs no register copies, so the backend emits
// only loads + WMMAs + partial s_wait_loadcnt in the steady state (no
// v_dual_mov rotation and none of the WAR-hazard v_nops they force).
// Requires K % 64 == 0 and K >= 128 (true for all uses: K in {256, 512}).
// ---------------------------------------------------------------------------
__global__ __launch_bounds__(128)
void gemm_wmma_kernel(const _Float16* __restrict__ A,
                      const _Float16* __restrict__ Bw,
                      const float* __restrict__ bias,
                      const float* __restrict__ residual,
                      float* __restrict__ Cf, _Float16* __restrict__ Ch,
                      int M, int N, int K, int relu) {
    const int lane = threadIdx.x & 31;
    const int w    = threadIdx.x >> 5;
    const int hi   = lane >> 4;
    const int lx   = lane & 15;
    const int mbase = blockIdx.y * 128 + w * 32;
    const int nbase = blockIdx.x * 64;

    // per-lane fragment base pointers (include row offset and hi*8 shift)
    const _Float16* pa0 = A + (size_t)(mbase + lx) * K + hi * 8;
    const _Float16* pa1 = A + (size_t)(mbase + 16 + lx) * K + hi * 8;
    const _Float16* pb[4];
#pragma unroll
    for (int nb = 0; nb < 4; ++nb)
        pb[nb] = Bw + (size_t)(nbase + nb * 16 + lx) * K + hi * 8;

    v8f acc[2][4];
#pragma unroll
    for (int i = 0; i < 2; ++i)
#pragma unroll
        for (int nb = 0; nb < 4; ++nb) acc[i][nb] = v8f{};

    // ping-pong fragment slots
    v16h A0a, A1a, Ba[4];
    v16h A0b, A1b, Bb[4];

    // preload slot A @ kk = 0
    A0a = frag_at(pa0, 0);
    A1a = frag_at(pa1, 0);
#pragma unroll
    for (int nb = 0; nb < 4; ++nb) Ba[nb] = frag_at(pb[nb], 0);

    int kk = 0;
    for (; kk + 64 < K; kk += 64) {
        // load slot B @ kk+32, compute slot A @ kk
        A0b = frag_at(pa0, kk + 32);
        A1b = frag_at(pa1, kk + 32);
#pragma unroll
        for (int nb = 0; nb < 4; ++nb) Bb[nb] = frag_at(pb[nb], kk + 32);
#pragma unroll
        for (int nb = 0; nb < 4; ++nb) {
            acc[0][nb] = wmma_f16(A0a, Ba[nb], acc[0][nb]);
            acc[1][nb] = wmma_f16(A1a, Ba[nb], acc[1][nb]);
        }
        // load slot A @ kk+64, compute slot B @ kk+32
        A0a = frag_at(pa0, kk + 64);
        A1a = frag_at(pa1, kk + 64);
#pragma unroll
        for (int nb = 0; nb < 4; ++nb) Ba[nb] = frag_at(pb[nb], kk + 64);
#pragma unroll
        for (int nb = 0; nb < 4; ++nb) {
            acc[0][nb] = wmma_f16(A0b, Bb[nb], acc[0][nb]);
            acc[1][nb] = wmma_f16(A1b, Bb[nb], acc[1][nb]);
        }
    }
    // tail: slot A holds kk (== K-64); load slot B @ K-32, compute both
    A0b = frag_at(pa0, kk + 32);
    A1b = frag_at(pa1, kk + 32);
#pragma unroll
    for (int nb = 0; nb < 4; ++nb) Bb[nb] = frag_at(pb[nb], kk + 32);
#pragma unroll
    for (int nb = 0; nb < 4; ++nb) {
        acc[0][nb] = wmma_f16(A0a, Ba[nb], acc[0][nb]);
        acc[1][nb] = wmma_f16(A1a, Ba[nb], acc[1][nb]);
    }
#pragma unroll
    for (int nb = 0; nb < 4; ++nb) {
        acc[0][nb] = wmma_f16(A0b, Bb[nb], acc[0][nb]);
        acc[1][nb] = wmma_f16(A1b, Bb[nb], acc[1][nb]);
    }

    // epilogue: bias / relu / residual, f32 and/or f16 stores
#pragma unroll
    for (int i = 0; i < 2; ++i) {
#pragma unroll
        for (int nb = 0; nb < 4; ++nb) {
            const int n = nbase + nb * 16 + lx;
            const float bn = bias ? bias[n] : 0.0f;
#pragma unroll
            for (int r = 0; r < 8; ++r) {
                const int m = mbase + i * 16 + r + hi * 8;  // lanes>=16 hold rows 8..15
                float v = acc[i][nb][r] + bn;
                if (relu) v = fmaxf(v, 0.0f);
                if (residual) v += residual[(size_t)m * N + n];
                if (Cf) Cf[(size_t)m * N + n] = v;
                if (Ch) Ch[(size_t)m * N + n] = (_Float16)v;
            }
        }
    }
}

// ---------------------------------------------------------------------------
// Flash-style attention. qkvh layout: [b, s, which(3), h, dk] f16 (row = 1536).
// vT layout: [b*h, dk, s] f16. heads out: [b, s, h*64+dk] f16.
// One block = one (b,h) and 64 query rows; each wave owns 16 rows.
// Online softmax: running max/sum live in registers (C-layout row groups
// align with lane halves), exp'ed P tile staged through 1KB/wave LDS to
// reshape into WMMA A-layout.
// ---------------------------------------------------------------------------
__global__ __launch_bounds__(128)
void attn_kernel(const _Float16* __restrict__ qkvh,
                 const _Float16* __restrict__ vT,
                 _Float16* __restrict__ heads) {
    __shared__ __align__(16) _Float16 ptile[4][16 * 32];
    const int lane = threadIdx.x & 31;
    const int w    = threadIdx.x >> 5;
    const int hi   = lane >> 4;
    const int lx   = lane & 15;
    const int bh = blockIdx.y;
    const int b = bh >> 3, h = bh & 7;
    const int mbase = blockIdx.x * 64 + w * 16;

    const _Float16* qbase = qkvh + (size_t)b * SS * 1536 + h * DK;          // which=0
    const _Float16* kbase = qkvh + (size_t)b * SS * 1536 + 512 + h * DK;    // which=1
    const _Float16* vtb   = vT + (size_t)bh * DK * SS;                      // [dk][t]
    _Float16* pt = ptile[w];

    // q fragments (K=64 -> kk = 0, 32), pre-scaled by 1/sqrt(DK)=0.125
    const int qrow = mbase + lx;
    v16h aq0 = load_frag(qbase, 1536, qrow, 0, hi);
    v16h aq1 = load_frag(qbase, 1536, qrow, 32, hi);
#pragma unroll
    for (int e = 0; e < 16; ++e) { aq0[e] = aq0[e] * (_Float16)0.125f;
                                   aq1[e] = aq1[e] * (_Float16)0.125f; }

    v8f oacc[4] = {v8f{}, v8f{}, v8f{}, v8f{}};
    float rmax[8], rsum[8];
#pragma unroll
    for (int r = 0; r < 8; ++r) { rmax[r] = -1e30f; rsum[r] = 0.0f; }

    for (int tb = 0; tb < SS; tb += 32) {
        // score tiles: s[j] covers key columns tb+j*16 .. tb+j*16+15
        v8f s0 = v8f{}, s1 = v8f{};
        {
            const int t0 = tb + lx, t1 = tb + 16 + lx;
            v16h bk;
            bk = load_frag(kbase, 1536, t0, 0, hi);  s0 = wmma_f16(aq0, bk, s0);
            bk = load_frag(kbase, 1536, t0, 32, hi); s0 = wmma_f16(aq1, bk, s0);
            bk = load_frag(kbase, 1536, t1, 0, hi);  s1 = wmma_f16(aq0, bk, s1);
            bk = load_frag(kbase, 1536, t1, 32, hi); s1 = wmma_f16(aq1, bk, s1);
        }
        // online softmax update; row groups: lanes 0-15 own rows 0..7, 16-31 own 8..15
#pragma unroll
        for (int r = 0; r < 8; ++r) {
            float tmax = fmaxf(s0[r], s1[r]);
            tmax = fmaxf(tmax, __shfl_xor(tmax, 1));
            tmax = fmaxf(tmax, __shfl_xor(tmax, 2));
            tmax = fmaxf(tmax, __shfl_xor(tmax, 4));
            tmax = fmaxf(tmax, __shfl_xor(tmax, 8));
            const float nmax = fmaxf(rmax[r], tmax);
            const float corr = __expf(rmax[r] - nmax);
            rmax[r] = nmax;
            const float p0 = __expf(s0[r] - nmax);
            const float p1 = __expf(s1[r] - nmax);
            float ts = p0 + p1;
            ts += __shfl_xor(ts, 1);
            ts += __shfl_xor(ts, 2);
            ts += __shfl_xor(ts, 4);
            ts += __shfl_xor(ts, 8);
            rsum[r] = rsum[r] * corr + ts;
#pragma unroll
            for (int nb = 0; nb < 4; ++nb) oacc[nb][r] *= corr;
            const int row = r + hi * 8;
            pt[row * 32 + lx]      = (_Float16)p0;
            pt[row * 32 + 16 + lx] = (_Float16)p1;
        }
        // reload P in A-layout and accumulate O += P * V
        v16h ap = load_frag(pt, 32, lx, 0, hi);
#pragma unroll
        for (int nb = 0; nb < 4; ++nb) {
            v16h bv = load_frag(vtb, SS, nb * 16 + lx, tb, hi);
            oacc[nb] = wmma_f16(ap, bv, oacc[nb]);
        }
    }
    // epilogue: divide by row sums, write heads in concat layout
#pragma unroll
    for (int nb = 0; nb < 4; ++nb) {
        const int n = h * DK + nb * 16 + lx;
#pragma unroll
        for (int r = 0; r < 8; ++r) {
            const int srow = mbase + r + hi * 8;
            heads[((size_t)b * SS + srow) * DD + n] =
                (_Float16)(oacc[nb][r] / rsum[r]);
        }
    }
}

// ---------------------------------------------------------------------------
// LayerNorm over the last (S,D) dims per batch. One block per batch sample.
// ---------------------------------------------------------------------------
__global__ __launch_bounds__(256)
void layernorm_kernel(const float* __restrict__ X,
                      const float* __restrict__ gamma,
                      const float* __restrict__ beta,
                      float* __restrict__ Y, _Float16* __restrict__ Yh) {
    __shared__ float sred[256], sred2[256];
    const size_t base = (size_t)blockIdx.x * LN_N;
    float s = 0.0f, s2 = 0.0f;
    for (int i = threadIdx.x; i < LN_N; i += 256) {
        const float v = X[base + i];
        s += v; s2 += v * v;
    }
    sred[threadIdx.x] = s; sred2[threadIdx.x] = s2;
    __syncthreads();
    for (int off = 128; off > 0; off >>= 1) {
        if (threadIdx.x < off) {
            sred[threadIdx.x]  += sred[threadIdx.x + off];
            sred2[threadIdx.x] += sred2[threadIdx.x + off];
        }
        __syncthreads();
    }
    const float mean = sred[0] * (1.0f / LN_N);
    const float var  = sred2[0] * (1.0f / LN_N) - mean * mean;
    const float rstd = rsqrtf(var + 1e-5f);
    for (int i = threadIdx.x; i < LN_N; i += 256) {
        const float v = (X[base + i] - mean) * rstd * gamma[i] + beta[i];
        Y[base + i] = v;
        if (Yh) Yh[base + i] = (_Float16)v;
    }
}

// ---------------------------------------------------------------------------
// Small data-prep kernels
// ---------------------------------------------------------------------------
__global__ void f32_to_f16_kernel(const float* __restrict__ src,
                                  _Float16* __restrict__ dst, int n) {
    const int i = blockIdx.x * 256 + threadIdx.x;
    if (i < n) dst[i] = (_Float16)src[i];
}

__global__ void pack_qkv_w_kernel(const float* __restrict__ Wq,
                                  const float* __restrict__ Wk,
                                  const float* __restrict__ Wv,
                                  _Float16* __restrict__ dst) {
    const int i = blockIdx.x * 256 + threadIdx.x;   // over 1536*512
    if (i >= 1536 * 512) return;
    const int n = i / 512, k = i % 512;
    const int which = n / 512, r = n % 512;          // r = h*DK + dk
    const float* W = (which == 0) ? Wq : (which == 1) ? Wk : Wv;
    dst[i] = (_Float16)W[r * 512 + k];
}

__global__ void pack_qkv_b_kernel(const float* __restrict__ bq,
                                  const float* __restrict__ bk,
                                  const float* __restrict__ bv,
                                  float* __restrict__ dst) {
    const int n = blockIdx.x * 256 + threadIdx.x;
    if (n >= 1536) return;
    const int which = n / 512, r = n % 512;
    const float* bb = (which == 0) ? bq : (which == 1) ? bk : bv;
    dst[n] = bb[r];
}

__global__ void convert_pad_kernel(const float* __restrict__ src,
                                   _Float16* __restrict__ dst,
                                   int Np, int Kp, int N, int K) {
    const int i = blockIdx.x * 256 + threadIdx.x;
    if (i >= Np * Kp) return;
    const int n = i / Kp, k = i % Kp;
    dst[i] = (n < N && k < K) ? (_Float16)src[n * K + k] : (_Float16)0.0f;
}

__global__ void pad_bias_kernel(const float* __restrict__ src,
                                float* __restrict__ dst, int Np, int N) {
    const int i = blockIdx.x * 256 + threadIdx.x;
    if (i < Np) dst[i] = (i < N) ? src[i] : 0.0f;
}

__global__ void transpose_v_kernel(const _Float16* __restrict__ qkvh,
                                   _Float16* __restrict__ vT) {
    const int i = blockIdx.x * 256 + threadIdx.x;    // over B*H*DK*S
    if (i >= BB * HH * DK * SS) return;
    const int t  = i & (SS - 1);
    const int dk = (i >> 9) & (DK - 1);
    const int bh = i >> 15;
    const int b = bh >> 3, h = bh & 7;
    vT[i] = qkvh[((size_t)(b * SS + t)) * 1536 + 1024 + h * DK + dk];
}

// ---------------------------------------------------------------------------
// Launch
// ---------------------------------------------------------------------------
extern "C" void kernel_launch(void* const* d_in, const int* in_sizes, int n_in,
                              void* d_out, int out_size, void* d_ws, size_t ws_size,
                              hipStream_t stream) {
    const float* x     = (const float*)d_in[0];
    const float* Wq    = (const float*)d_in[1];
    const float* bq    = (const float*)d_in[2];
    const float* Wk    = (const float*)d_in[3];
    const float* bk    = (const float*)d_in[4];
    const float* Wv    = (const float*)d_in[5];
    const float* bv    = (const float*)d_in[6];
    const float* Wo    = (const float*)d_in[7];
    const float* bo    = (const float*)d_in[8];
    const float* W1    = (const float*)d_in[9];
    const float* b1    = (const float*)d_in[10];
    const float* W2    = (const float*)d_in[11];
    const float* b2    = (const float*)d_in[12];
    const float* W3    = (const float*)d_in[13];
    const float* b3    = (const float*)d_in[14];
    const float* gamma = (const float*)d_in[15];
    const float* beta  = (const float*)d_in[16];
    float* out = (float*)d_out;

    // workspace carve-up (256B-aligned so v8h loads stay 16B-aligned)
    char* base = (char*)d_ws;
    size_t off = 0;
    auto carve = [&](size_t bytes) {
        void* p = base + off;
        off += (bytes + 255) & ~(size_t)255;
        return p;
    };
    const size_t BSD = (size_t)MTOT * DD;   // 8,388,608
    _Float16* xh     = (_Float16*)carve(BSD * 2);
    _Float16* qkvh   = (_Float16*)carve(BSD * 3 * 2);
    _Float16* vT     = (_Float16*)carve(BSD * 2);
    _Float16* headsh = (_Float16*)carve(BSD * 2);
    float*    y1     = (float*)   carve(BSD * 4);
    _Float16* y1h    = (_Float16*)carve(BSD * 2);
    _Float16* f1h    = (_Float16*)carve((size_t)MTOT * FFP * 2);
    _Float16* f2h    = (_Float16*)carve((size_t)MTOT * FFP * 2);
    float*    y2     = (float*)   carve(BSD * 4);
    _Float16* Wqkvh  = (_Float16*)carve((size_t)1536 * 512 * 2);
    _Float16* Woh    = (_Float16*)carve((size_t)512 * 512 * 2);
    _Float16* W1h    = (_Float16*)carve((size_t)FFP * 512 * 2);
    _Float16* W2h    = (_Float16*)carve((size_t)FFP * FFP * 2);
    _Float16* W3h    = (_Float16*)carve((size_t)512 * FFP * 2);
    float*    bqkv   = (float*)   carve(1536 * 4);
    float*    b1p    = (float*)   carve(FFP * 4);
    float*    b2p    = (float*)   carve(FFP * 4);
    (void)ws_size; (void)in_sizes; (void)n_in; (void)out_size;

    // --- data prep ---
    f32_to_f16_kernel<<<(int)((BSD + 255) / 256), 256, 0, stream>>>(x, xh, (int)BSD);
    pack_qkv_w_kernel<<<(1536 * 512 + 255) / 256, 256, 0, stream>>>(Wq, Wk, Wv, Wqkvh);
    pack_qkv_b_kernel<<<(1536 + 255) / 256, 256, 0, stream>>>(bq, bk, bv, bqkv);
    convert_pad_kernel<<<(512 * 512 + 255) / 256, 256, 0, stream>>>(Wo, Woh, 512, 512, 512, 512);
    convert_pad_kernel<<<(FFP * 512 + 255) / 256, 256, 0, stream>>>(W1, W1h, FFP, 512, FF, 512);
    convert_pad_kernel<<<(FFP * FFP + 255) / 256, 256, 0, stream>>>(W2, W2h, FFP, FFP, FF, FF);
    convert_pad_kernel<<<(512 * FFP + 255) / 256, 256, 0, stream>>>(W3, W3h, 512, FFP, 512, FF);
    pad_bias_kernel<<<1, 256, 0, stream>>>(b1, b1p, FFP, FF);
    pad_bias_kernel<<<1, 256, 0, stream>>>(b2, b2p, FFP, FF);

    // --- QKV projection: [16384,512] x [1536,512]^T -> qkvh f16 ---
    gemm_wmma_kernel<<<dim3(1536 / 64, MTOT / 128), 128, 0, stream>>>(
        xh, Wqkvh, bqkv, nullptr, nullptr, qkvh, MTOT, 1536, 512, 0);

    // --- transpose V for contiguous B-fragment loads ---
    transpose_v_kernel<<<(int)((BSD + 255) / 256), 256, 0, stream>>>(qkvh, vT);

    // --- attention ---
    attn_kernel<<<dim3(SS / 64, BB * HH), 128, 0, stream>>>(qkvh, vT, headsh);

    // --- output projection + residual(x) -> y1 f32 ---
    gemm_wmma_kernel<<<dim3(DD / 64, MTOT / 128), 128, 0, stream>>>(
        headsh, Woh, bo, x, y1, nullptr, MTOT, DD, 512, 0);

    // --- LN1 (in-place, also emit f16 for FFN) ---
    layernorm_kernel<<<BB, 256, 0, stream>>>(y1, gamma, beta, y1, y1h);

    // --- FFN ---
    gemm_wmma_kernel<<<dim3(FFP / 64, MTOT / 128), 128, 0, stream>>>(
        y1h, W1h, b1p, nullptr, nullptr, f1h, MTOT, FFP, 512, 1);
    gemm_wmma_kernel<<<dim3(FFP / 64, MTOT / 128), 128, 0, stream>>>(
        f1h, W2h, b2p, nullptr, nullptr, f2h, MTOT, FFP, FFP, 1);
    gemm_wmma_kernel<<<dim3(DD / 64, MTOT / 128), 128, 0, stream>>>(
        f2h, W3h, b3, y1, y2, nullptr, MTOT, DD, FFP, 0);

    // --- LN2 (in-place), LN3 -> output ---
    layernorm_kernel<<<BB, 256, 0, stream>>>(y2, gamma, beta, y2, nullptr);
    layernorm_kernel<<<BB, 256, 0, stream>>>(y2, gamma, beta, out, nullptr);
}
